// TopKSAE_2954937499787
// MI455X (gfx1250) — compile-verified, matrix operand order checked by reading, not run on hardware
//
#include <hip/hip_runtime.h>

typedef __attribute__((ext_vector_type(16))) __bf16 bf16x16;
typedef __attribute__((ext_vector_type(8)))  __bf16 bf16x8;
typedef __attribute__((ext_vector_type(8)))  float  f32x8;
typedef __attribute__((ext_vector_type(4)))  float  f32x4;

#define D_IN   2048
#define D_SAE  32768
#define BATCH  4096
#define KMAX   256

#define BM 128
#define BN 128
#define BK 64
#define SAe 72   // padded LDS row stride (bf16 elems) for A tile [row][k]
#define SBe 72   // padded LDS row stride (bf16 elems) for B tile [col][k]

// ---------------------------------------------------------------------------
// 1) xc = bf16(x - b_dec)
// ---------------------------------------------------------------------------
__global__ void k_xc(const float* __restrict__ x, const float* __restrict__ b_dec,
                     __bf16* __restrict__ xc) {
  size_t i = (size_t)blockIdx.x * blockDim.x + threadIdx.x;
  const size_t total = (size_t)BATCH * D_IN;
  const size_t step = (size_t)gridDim.x * blockDim.x;
  for (; i < total; i += step)
    xc[i] = (__bf16)(x[i] - b_dec[i & (D_IN - 1)]);
}

// ---------------------------------------------------------------------------
// 2) dec_norms and 1/dec_norms (one wave per W_dec row)
// ---------------------------------------------------------------------------
__global__ void k_norms(const float* __restrict__ Wd, float* __restrict__ dn,
                        float* __restrict__ inv) {
  const int wave = threadIdx.x >> 5, lane = threadIdx.x & 31;
  const int row = blockIdx.x * 8 + wave;
  const float* p = Wd + (size_t)row * D_IN;
  float s = 0.f;
  for (int i = lane; i < D_IN; i += 32) { float v = p[i]; s = fmaf(v, v, s); }
  for (int o = 16; o; o >>= 1) s += __shfl_xor(s, o, 32);
  if (lane == 0) {
    float n = sqrtf(s);
    dn[row] = n;
    inv[row] = (n > 0.f) ? (1.f / n) : 0.f;
  }
}

// ---------------------------------------------------------------------------
// 3) Encode GEMM: hidden = relu(((x-b_dec) @ W_enc + b_enc) * dec_norms)
//    bf16 WMMA 16x16x32, f32 accumulate. Block tile 128x128, K-step 64
//    (2 WMMA K-slices per stage => 16 WMMAs per barrier pair).
//    A tile staged via GLOBAL_LOAD_ASYNC_TO_LDS_B128 (ASYNCcnt path).
//    8 waves: 2(M) x 4(N), each wave owns a 64x32 patch = 4x2 accumulators.
// ---------------------------------------------------------------------------
__global__ __launch_bounds__(256) void k_encode(
    const __bf16* __restrict__ xc, const float* __restrict__ We,
    const float* __restrict__ b_enc, const float* __restrict__ dn,
    float* __restrict__ hidden) {
  __shared__ __bf16 sA[BM * SAe];
  __shared__ __bf16 sB[BN * SBe];

  const int tid = threadIdx.x;
  const int wave = tid >> 5, lane = tid & 31;
  const int wm = wave >> 2;         // 0..1 : 64-row strip
  const int wn = wave & 3;          // 0..3 : 32-col strip
  const int h = lane >> 4;          // half-lane select (K-halves per ISA layout)
  const int r = lane & 15;
  const int m0 = blockIdx.x * BM;
  const int n0 = blockIdx.y * BN;

  f32x8 acc[4][2];
#pragma unroll
  for (int mt = 0; mt < 4; ++mt)
#pragma unroll
    for (int nt = 0; nt < 2; ++nt)
#pragma unroll
      for (int i = 0; i < 8; ++i) acc[mt][nt][i] = 0.f;

  // A loader: 128 rows x 64 bf16 = 16 KB; each thread: 4 async b128 (16B each)
  const int arow = tid >> 1, aseg = (tid & 1) * 4;
  // B loader: 64 k-rows x 128 cols f32; thread owns k-pair (2*kp, 2*kp+1) x 16 cols
  const int kp = tid & 31, colseg = (tid >> 5) * 16;

  for (int k0 = 0; k0 < D_IN; k0 += BK) {
    // --- A tile: bf16 global -> LDS, asynchronous (no VGPR data path) ---
#pragma unroll
    for (int j = 0; j < 4; ++j) {
      const void* gsrc = (const void*)(xc + (size_t)(m0 + arow) * D_IN + k0 + (aseg + j) * 8);
      unsigned ldsoff = (unsigned)(uintptr_t)(void*)&sA[arow * SAe + (aseg + j) * 8];
      asm volatile("global_load_async_to_lds_b128 %0, %1, off"
                   :: "v"(ldsoff), "v"(gsrc) : "memory");
    }

    // --- B tile: f32 global -> VGPR -> packed bf16 pairs -> LDS [col][k] ---
    const float* gb0 = We + (size_t)(k0 + 2 * kp) * D_SAE + n0 + colseg;
    const float* gb1 = gb0 + D_SAE;
    f32x4 ra[4], rb[4];
#pragma unroll
    for (int j = 0; j < 4; ++j) {
      ra[j] = ((const f32x4*)gb0)[j];
      rb[j] = ((const f32x4*)gb1)[j];
    }
    if (k0 + BK < D_IN) __builtin_prefetch(gb0 + (size_t)BK * D_SAE, 0, 1);

#pragma unroll
    for (int j = 0; j < 16; ++j) {
      union { __bf16 hh[2]; unsigned u; } pk;
      pk.hh[0] = (__bf16)ra[j >> 2][j & 3];   // k = 2*kp
      pk.hh[1] = (__bf16)rb[j >> 2][j & 3];   // k = 2*kp+1
      *(unsigned*)&sB[(colseg + j) * SBe + 2 * kp] = pk.u;
    }

    // async A writes must land in LDS before the workgroup barrier completes
    asm volatile("s_wait_asynccnt 0" ::: "memory");
    __syncthreads();

    // --- 2 K-slices of fragments + WMMAs ---
#pragma unroll
    for (int s = 0; s < 2; ++s) {
      bf16x16 afr[4], bfr[2];
#pragma unroll
      for (int mt = 0; mt < 4; ++mt) {
        const int row = wm * 64 + mt * 16 + r;
        bf16x8 lo = *(const bf16x8*)&sA[row * SAe + s * 32 + h * 8];
        bf16x8 hi = *(const bf16x8*)&sA[row * SAe + s * 32 + 16 + h * 8];
#pragma unroll
        for (int i = 0; i < 8; ++i) { afr[mt][i] = lo[i]; afr[mt][8 + i] = hi[i]; }
      }
#pragma unroll
      for (int nt = 0; nt < 2; ++nt) {
        const int col = wn * 32 + nt * 16 + r;
        bf16x8 lo = *(const bf16x8*)&sB[col * SBe + s * 32 + h * 8];
        bf16x8 hi = *(const bf16x8*)&sB[col * SBe + s * 32 + 16 + h * 8];
#pragma unroll
        for (int i = 0; i < 8; ++i) { bfr[nt][i] = lo[i]; bfr[nt][8 + i] = hi[i]; }
      }
#pragma unroll
      for (int mt = 0; mt < 4; ++mt)
#pragma unroll
        for (int nt = 0; nt < 2; ++nt)
          acc[mt][nt] = __builtin_amdgcn_wmma_f32_16x16x32_bf16(
              false, afr[mt], false, bfr[nt], (short)0, acc[mt][nt], false, false);
    }
    __syncthreads();
  }

  // --- fused epilogue: (acc + b_enc) * dec_norm, ReLU, store dense hidden ---
#pragma unroll
  for (int nt = 0; nt < 2; ++nt) {
    const int n = n0 + wn * 32 + nt * 16 + r;
    const float d = dn[n];
    const float be = b_enc[n];
#pragma unroll
    for (int mt = 0; mt < 4; ++mt) {
#pragma unroll
      for (int i = 0; i < 8; ++i) {
        const int m = m0 + wm * 64 + mt * 16 + h * 8 + i;  // C/D: lanes 16-31 hold M=8..15
        float v = (acc[mt][nt][i] + be) * d;
        hidden[(size_t)m * D_SAE + n] = v > 0.f ? v : 0.f;
      }
    }
  }
}

// ---------------------------------------------------------------------------
// 4) Per-row top-k via radix select (values are >=0 so uint view is monotonic)
//    In-place sparsify + deterministic compaction of (idx, val) lists.
// ---------------------------------------------------------------------------
__global__ __launch_bounds__(256) void k_topk(float* __restrict__ hidden,
                                              const int* __restrict__ kptr,
                                              float* __restrict__ tv,
                                              int* __restrict__ ti,
                                              int* __restrict__ tc) {
  __shared__ unsigned hist[256];
  __shared__ unsigned s_gt[257];
  __shared__ unsigned s_eq[257];
  __shared__ unsigned sh_prefix, sh_need;

  const int tid = threadIdx.x;
  const int row = blockIdx.x;
  float* p = hidden + (size_t)row * D_SAE;

  int kk = *kptr;
  if (kk < 1) kk = 1;
  if (kk > KMAX) kk = KMAX;

  if (tid == 0) { sh_prefix = 0u; sh_need = (unsigned)kk; }
  __syncthreads();

  // 4 radix passes, 8 bits each, high digit first
  for (int pass = 0; pass < 4; ++pass) {
    const int shift = 24 - 8 * pass;
    const unsigned himask = pass ? (0xFFFFFFFFu << (shift + 8)) : 0u;
    const unsigned prefix = sh_prefix;
    hist[tid] = 0u;
    __syncthreads();
    for (int e = tid; e < D_SAE; e += 256) {
      unsigned key = __float_as_uint(p[e]);
      if ((key & himask) == prefix)
        atomicAdd(&hist[(key >> shift) & 255], 1u);
    }
    __syncthreads();
    if (tid == 0) {
      unsigned need = sh_need, cum = 0; int b = 0;
      for (int d = 255; d >= 0; --d) {
        unsigned c = hist[d];
        if (cum + c >= need) { b = d; break; }
        cum += c;
      }
      sh_prefix = prefix | ((unsigned)b << shift);
      sh_need = need - cum;
    }
    __syncthreads();
  }

  const unsigned T = sh_prefix;       // k-th largest key (exact)
  const unsigned needT = sh_need;     // how many ties at T to keep

  // deterministic compaction: each thread owns a contiguous chunk of 128
  const int base = tid * (D_SAE / 256);
  unsigned cgt = 0, ceq = 0;
  for (int i = 0; i < D_SAE / 256; ++i) {
    unsigned key = __float_as_uint(p[base + i]);
    cgt += (key > T);
    ceq += (key == T);
  }
  s_gt[tid] = cgt; s_eq[tid] = ceq;
  __syncthreads();
  if (tid == 0) {
    unsigned ag = 0, ae = 0;
    for (int i = 0; i < 256; ++i) {
      unsigned g = s_gt[i], e = s_eq[i];
      s_gt[i] = ag; s_eq[i] = ae;
      ag += g; ae += e;
    }
    s_gt[256] = ag; s_eq[256] = ae;
  }
  __syncthreads();

  unsigned slot_gt = s_gt[tid];
  unsigned rank_eq = s_eq[tid];
  const unsigned total_gt = s_gt[256];
  const unsigned total_eq = s_eq[256];

  for (int i = 0; i < D_SAE / 256; ++i) {
    const int e = base + i;
    const float v = p[e];
    const unsigned key = __float_as_uint(v);
    bool keep = false; unsigned s = 0;
    if (key > T) { keep = true; s = slot_gt++; }
    else if (key == T) {
      unsigned rk = rank_eq++;
      if (rk < needT) { keep = true; s = total_gt + rk; }
    }
    if (keep) {
      tv[(size_t)row * KMAX + s] = v;
      ti[(size_t)row * KMAX + s] = e;
    } else {
      p[e] = 0.f;
    }
  }
  if (tid == 0) {
    unsigned take = needT < total_eq ? needT : total_eq;
    tc[row] = (int)(total_gt + take);
  }
}

// ---------------------------------------------------------------------------
// 5) Sparse decode: recon = sum_j (val_j / ||dec_j||) * W_dec[idx_j] + b_dec
//    + fused per-row squared error.
// ---------------------------------------------------------------------------
__global__ __launch_bounds__(256) void k_decode(
    const float* __restrict__ tv, const int* __restrict__ ti,
    const int* __restrict__ tc, const float* __restrict__ inv,
    const float* __restrict__ Wd, const float* __restrict__ b_dec,
    const float* __restrict__ x, float* __restrict__ recon,
    float* __restrict__ row_sq) {
  __shared__ float sc[KMAX];
  __shared__ int si[KMAX];
  __shared__ float swsum[8];

  const int tid = threadIdx.x;
  const int row = blockIdx.x;
  const int cnt = tc[row];

  if (tid < cnt) {
    int idx = ti[(size_t)row * KMAX + tid];
    si[tid] = idx;
    sc[tid] = tv[(size_t)row * KMAX + tid] * inv[idx];
  }
  __syncthreads();

  float acc[8];
#pragma unroll
  for (int c = 0; c < 8; ++c) acc[c] = 0.f;

  for (int j = 0; j < cnt; ++j) {
    const float coef = sc[j];
    const float* wr = Wd + (size_t)si[j] * D_IN;
#pragma unroll
    for (int c = 0; c < 8; ++c)
      acc[c] = fmaf(coef, wr[tid + 256 * c], acc[c]);
  }

  const float* xr = x + (size_t)row * D_IN;
  float* rr = recon + (size_t)row * D_IN;
  float sq = 0.f;
#pragma unroll
  for (int c = 0; c < 8; ++c) {
    const int col = tid + 256 * c;
    float v = acc[c] + b_dec[col];
    rr[col] = v;
    float d = v - xr[col];
    sq = fmaf(d, d, sq);
  }
  for (int o = 16; o; o >>= 1) sq += __shfl_xor(sq, o, 32);
  if ((tid & 31) == 0) swsum[tid >> 5] = sq;
  __syncthreads();
  if (tid == 0) {
    float s = 0.f;
    for (int w = 0; w < 8; ++w) s += swsum[w];
    row_sq[row] = s;
  }
}

// ---------------------------------------------------------------------------
// 6) Final loss reduction (deterministic)
// ---------------------------------------------------------------------------
__global__ void k_loss(const float* __restrict__ row_sq, float* __restrict__ out) {
  __shared__ float sw[8];
  const int tid = threadIdx.x;
  float s = 0.f;
  for (int i = tid; i < BATCH; i += 256) s += row_sq[i];
  for (int o = 16; o; o >>= 1) s += __shfl_xor(s, o, 32);
  if ((tid & 31) == 0) sw[tid >> 5] = s;
  __syncthreads();
  if (tid == 0) {
    float t = 0.f;
    for (int w = 0; w < 8; ++w) t += sw[w];
    float mse = t / (float)BATCH;
    out[0] = mse;   // loss = mse + aux(0)
    out[1] = mse;   // mse_loss
    out[2] = 0.f;   // aux_loss
  }
}

// ---------------------------------------------------------------------------
extern "C" void kernel_launch(void* const* d_in, const int* in_sizes, int n_in,
                              void* d_out, int out_size, void* d_ws, size_t ws_size,
                              hipStream_t stream) {
  (void)in_sizes; (void)n_in; (void)out_size; (void)ws_size;

  const float* x   = (const float*)d_in[0];
  const float* We  = (const float*)d_in[1];
  const float* be  = (const float*)d_in[2];
  const float* Wd  = (const float*)d_in[3];
  const float* bd  = (const float*)d_in[4];
  const int*  kptr = (const int*)d_in[5];

  float* out    = (float*)d_out;
  float* recon  = out;                                   // [BATCH, D_IN]
  float* sparse = out + (size_t)BATCH * D_IN;            // [BATCH, D_SAE]
  float* scal   = sparse + (size_t)BATCH * D_SAE;        // [loss, mse, aux]

  char* ws = (char*)d_ws;
  __bf16* xc   = (__bf16*)ws;  ws += (size_t)BATCH * D_IN * sizeof(__bf16);
  float* dn    = (float*)ws;   ws += (size_t)D_SAE * sizeof(float);
  float* inv   = (float*)ws;   ws += (size_t)D_SAE * sizeof(float);
  float* tv    = (float*)ws;   ws += (size_t)BATCH * KMAX * sizeof(float);
  int*   ti    = (int*)ws;     ws += (size_t)BATCH * KMAX * sizeof(int);
  int*   tc    = (int*)ws;     ws += (size_t)BATCH * sizeof(int);
  float* rowsq = (float*)ws;   ws += (size_t)BATCH * sizeof(float);

  k_xc<<<4096, 256, 0, stream>>>(x, bd, xc);
  k_norms<<<D_SAE / 8, 256, 0, stream>>>(Wd, dn, inv);

  dim3 g(BATCH / BM, D_SAE / BN);   // M fast-varying -> W_enc tile reuse via L2
  k_encode<<<g, 256, 0, stream>>>(xc, We, be, dn, sparse);

  k_topk<<<BATCH, 256, 0, stream>>>(sparse, kptr, tv, ti, tc);
  k_decode<<<BATCH, 256, 0, stream>>>(tv, ti, tc, inv, Wd, bd, x, recon, rowsq);
  k_loss<<<1, 256, 0, stream>>>(rowsq, scal);
}
